// DandCrecon_72799695667792
// MI455X (gfx1250) — matrix operand order repaired and checked by reading, not compiled, not run on hardware
//
#include <hip/hip_runtime.h>
#include <hip/hip_bf16.h>

typedef __attribute__((ext_vector_type(16))) _Float16 v16h;
typedef __attribute__((ext_vector_type(8)))  float    v8f;
typedef __attribute__((ext_vector_type(4)))  int      v4i;

#define N_PROJ 96
#define N_ANG  96
#define N_DET  384
#define N_U    384
#define N_V    384
#define VOLD   192
#define PI_F   3.14159265358979323846f

#if __has_builtin(__builtin_amdgcn_global_load_async_to_lds_b128) && \
    __has_builtin(__builtin_amdgcn_s_wait_asynccnt)
#define USE_ASYNC_LDS 1
typedef __attribute__((address_space(1))) v4i glob_v4i;   // global int4*
typedef __attribute__((address_space(3))) v4i lds_v4i;    // LDS int4*
#else
#define USE_ASYNC_LDS 0
#endif

// ---------------------------------------------------------------------------
// Kernel 1: s = gradient(sino * weight, axis=det) as a banded WMMA matmul.
// G (384x384) is tridiagonal-banded, so a 16-col output tile only needs
// detector rows [n0-1, n0+16]: ONE 32-wide K window -> ONE v_wmma per tile.
// Each wave computes two 16x16 row-tiles sharing a single B fragment.
// B is synthesized branchlessly (v_cndmask chains, no exec divergence:
// WMMA requires EXEC all-ones).
// ---------------------------------------------------------------------------
__global__ __launch_bounds__(256) void grad_wmma_kernel(
    const float* __restrict__ sino,   // [9216 x 384]
    const float* __restrict__ wgt,    // [9216 x 384]
    float* __restrict__ g)            // [9216 x 384]
{
    const int wave = blockIdx.x * (blockDim.x >> 5) + (threadIdx.x >> 5);
    const int lane = threadIdx.x & 31;
    const int TILE_COLS = N_DET / 16;          // 24
    const int tile_r = wave / TILE_COLS;       // 0..287 (two 16-row tiles each)
    const int tile_c = wave % TILE_COLS;
    const int r0 = tile_r * 32;
    const int n0 = tile_c * 16;

    const int khi  = lane >> 4;                // half-wave id
    const int jcol = n0 + (lane & 15);         // output detector column

    // 32-wide K window covering the band [n0-1, n0+16]
    int kw = n0 - 8;
    kw = kw < 0 ? 0 : (kw > N_DET - 32 ? N_DET - 32 : kw);

    // ---- B fragment (32x16 K x N): lanes 0-15 K rows kw+{0..7,16..23},
    //      lanes 16-31 K rows kw+{8..15,24..31}; column = lane&15. Branchless.
    const int kb = kw + 8 * khi;
    v16h b;
#pragma unroll
    for (int h = 0; h < 16; ++h) {
        const int i = kb + (h < 8 ? h : h + 8);
        float gi = 0.0f;                       // interior central difference
        gi = (i == jcol - 1) ? -0.5f : gi;
        gi = (i == jcol + 1) ?  0.5f : gi;
        float gl = 0.0f;                       // left edge (jcol==0)
        gl = (i == 0) ? -1.0f : gl;
        gl = (i == 1) ?  1.0f : gl;
        float gr = 0.0f;                       // right edge (jcol==383)
        gr = (i == N_DET - 2) ? -1.0f : gr;
        gr = (i == N_DET - 1) ?  1.0f : gr;
        float gv = gi;
        gv = (jcol == 0)         ? gl : gv;
        gv = (jcol == N_DET - 1) ? gr : gv;
        b[h] = (_Float16)gv;
    }

    // ---- A fragments for the two row-tiles (16-bit A 16x32 layout:
    // lanes 0-15 hold K=kw+{0..7,16..23}; lanes 16-31 K=kw+{8..15,24..31})
    const int ka = kw + khi * 8;
    v8f acc0 = {};
    v8f acc1 = {};

#pragma unroll
    for (int t = 0; t < 2; ++t) {
        const int row = r0 + t * 16 + (lane & 15);
        const float* __restrict__ srow = sino + (size_t)row * N_DET;
        const float* __restrict__ wrow = wgt  + (size_t)row * N_DET;
        const float4 s0 = *(const float4*)(srow + ka);
        const float4 s1 = *(const float4*)(srow + ka + 4);
        const float4 s2 = *(const float4*)(srow + ka + 16);
        const float4 s3 = *(const float4*)(srow + ka + 20);
        const float4 w0 = *(const float4*)(wrow + ka);
        const float4 w1 = *(const float4*)(wrow + ka + 4);
        const float4 w2 = *(const float4*)(wrow + ka + 16);
        const float4 w3 = *(const float4*)(wrow + ka + 20);

        v16h a;
        a[0]  = (_Float16)(s0.x * w0.x);  a[1]  = (_Float16)(s0.y * w0.y);
        a[2]  = (_Float16)(s0.z * w0.z);  a[3]  = (_Float16)(s0.w * w0.w);
        a[4]  = (_Float16)(s1.x * w1.x);  a[5]  = (_Float16)(s1.y * w1.y);
        a[6]  = (_Float16)(s1.z * w1.z);  a[7]  = (_Float16)(s1.w * w1.w);
        a[8]  = (_Float16)(s2.x * w2.x);  a[9]  = (_Float16)(s2.y * w2.y);
        a[10] = (_Float16)(s2.z * w2.z);  a[11] = (_Float16)(s2.w * w2.w);
        a[12] = (_Float16)(s3.x * w3.x);  a[13] = (_Float16)(s3.y * w3.y);
        a[14] = (_Float16)(s3.z * w3.z);  a[15] = (_Float16)(s3.w * w3.w);

        if (t == 0)
            acc0 = __builtin_amdgcn_wmma_f32_16x16x32_f16(
                false, a, false, b, (short)0, acc0, false, false);
        else
            acc1 = __builtin_amdgcn_wmma_f32_16x16x32_f16(
                false, a, false, b, (short)0, acc1, false, false);
    }

    // ---- store D: VGPR v holds M = v + 8*khi, N = lane&15
#pragma unroll
    for (int v = 0; v < 8; ++v) {
        const int m = v + 8 * khi;
        g[(size_t)(r0 + m) * N_DET + jcol]      = acc0[v];
        g[(size_t)(r0 + 16 + m) * N_DET + jcol] = acc1[v];
    }
}

// ---------------------------------------------------------------------------
// Kernel 2: 2-D parallel-beam backprojection + weight_3d.
// Double-buffered LDS staging of 16-angle sinogram chunks (2 x 24 KB out of
// 320 KB/WGP) using gfx1250 async global->LDS DMA: the next chunk's DMA
// overlaps the current chunk's lerp/gather math (tracked via ASYNCcnt).
// ---------------------------------------------------------------------------
__global__ __launch_bounds__(256) void bp2d_kernel(
    const float* __restrict__ g,      // [96][96][384]
    const float* __restrict__ w3d,    // [384][384]
    float* __restrict__ proj)         // [96][384][384]
{
    __shared__ float lds_s[2][16 * N_DET];
    __shared__ float lds_trig[2 * N_ANG];

    const int b  = blockIdx.z;
    const int tx = threadIdx.x & 15;
    const int ty = threadIdx.x >> 4;
    const int x  = blockIdx.x * 16 + tx;
    const int y  = blockIdx.y * 16 + ty;

    if (threadIdx.x < N_ANG) {
        float th = (float)threadIdx.x * (PI_F / (float)N_ANG);
        float sn, cs;
        __sincosf(th, &sn, &cs);
        lds_trig[2 * threadIdx.x + 0] = cs;
        lds_trig[2 * threadIdx.x + 1] = sn;
    }

    const float* __restrict__ gb = g + (size_t)b * N_ANG * N_DET;

#if USE_ASYNC_LDS
    // async-DMA one 16-angle chunk (6144 floats = 1536 b128) into buffer `buf`
    auto stage_async = [&](int a0, int buf) {
        const float* src = gb + (size_t)a0 * N_DET;
        for (int i = threadIdx.x; i < (16 * N_DET) / 4; i += 256) {
            __builtin_amdgcn_global_load_async_to_lds_b128(
                (glob_v4i*)(src + i * 4),
                (lds_v4i*)(&lds_s[buf][i * 4]), 0, 0);
        }
    };
    stage_async(0, 0);
    __builtin_amdgcn_s_wait_asynccnt(0);
    __syncthreads();
#else
    for (int i = threadIdx.x; i < 16 * N_DET; i += 256)
        lds_s[0][i] = gb[i];
    __syncthreads();
#endif

    const float xx = (float)x - 191.5f;   // (x - (N_U-1)/2) * PIX
    const float yy = (float)y - 191.5f;
    float acc = 0.0f;

    for (int c = 0; c < N_ANG / 16; ++c) {
        const int cur = c & 1;
#if USE_ASYNC_LDS
        if (c + 1 < N_ANG / 16)
            stage_async((c + 1) * 16, cur ^ 1);   // overlap DMA with compute
#endif
#pragma unroll 4
        for (int aa = 0; aa < 16; ++aa) {
            const int a = c * 16 + aa;
            const float cs = lds_trig[2 * a + 0];
            const float sn = lds_trig[2 * a + 1];
            const float si  = xx * cs + yy * sn + 191.5f;
            const float i0f = floorf(si);
            const float w   = si - i0f;
            const int i0 = (int)i0f;
            const int i1 = i0 + 1;
            const float v0 = (i0 >= 0 && i0 < N_DET) ? lds_s[cur][aa * N_DET + i0] : 0.0f;
            const float v1 = (i1 >= 0 && i1 < N_DET) ? lds_s[cur][aa * N_DET + i1] : 0.0f;
            acc += (1.0f - w) * v0 + w * v1;
        }
#if USE_ASYNC_LDS
        __builtin_amdgcn_s_wait_asynccnt(0);
        __syncthreads();
#else
        __syncthreads();
        if (c + 1 < N_ANG / 16)
            for (int i = threadIdx.x; i < 16 * N_DET; i += 256)
                lds_s[cur ^ 1][i] = gb[(size_t)(c + 1) * 16 * N_DET + i];
        __syncthreads();
#endif
    }

    const float scale = PI_F / (float)N_ANG;
    proj[((size_t)b * N_V + y) * N_U + x] = acc * scale * w3d[y * N_U + x];
}

// ---------------------------------------------------------------------------
// Kernel 3: 3-D cone-beam backprojection + ReLU. proj (57 MB) stays resident
// in the 192 MB L2 — direct bilinear gathers; LDS sincos; prefetch hints.
// ---------------------------------------------------------------------------
__global__ __launch_bounds__(256) void bp3d_kernel(
    const float* __restrict__ proj,   // [96][384][384]
    float* __restrict__ out)          // [192][192][192]
{
    __shared__ float trig[2 * N_PROJ];

    const int tx = threadIdx.x & 15;
    const int ty = threadIdx.x >> 4;
    const int x  = blockIdx.x * 16 + tx;
    const int y  = blockIdx.y * 16 + ty;
    const int z  = blockIdx.z;

    if (threadIdx.x < N_PROJ) {
        float be = (float)threadIdx.x * (2.0f * PI_F / (float)N_PROJ);
        float sn, cs;
        __sincosf(be, &sn, &cs);
        trig[2 * threadIdx.x + 0] = cs;
        trig[2 * threadIdx.x + 1] = sn;
    }
    __syncthreads();

    const float Xc = ((float)x - 95.5f) * 0.5f;   // (i - (VOL-1)/2) * VOX
    const float Yc = ((float)y - 95.5f) * 0.5f;
    const float Zc = ((float)z - 95.5f) * 0.5f;

    float acc = 0.0f;

    for (int p = 0; p < N_PROJ; ++p) {
        const float cb = trig[2 * p + 0];
        const float sb = trig[2 * p + 1];
        const float xr =  Xc * cb + Yc * sb;
        const float yr = -Xc * sb + Yc * cb;
        const float mag = 1200.0f / (750.0f - yr);   // SDD / (SID - yr)
        const float ui = xr * mag + 191.5f;
        const float vi = Zc * mag + 191.5f;

        const float u0f = floorf(ui), v0f = floorf(vi);
        const float wu = ui - u0f,   wv = vi - v0f;
        const int u0 = (int)u0f, v0 = (int)v0f;
        const int u1 = u0 + 1,   v1 = v0 + 1;

        const bool uok0 = (u0 >= 0) & (u0 < N_U);
        const bool uok1 = (u1 >= 0) & (u1 < N_U);
        const bool vok0 = (v0 >= 0) & (v0 < N_V);
        const bool vok1 = (v1 >= 0) & (v1 < N_V);
        const int cu0 = min(max(u0, 0), N_U - 1);
        const int cu1 = min(max(u1, 0), N_U - 1);
        const int cv0 = min(max(v0, 0), N_V - 1);
        const int cv1 = min(max(v1, 0), N_V - 1);

        const float* __restrict__ pp = proj + (size_t)p * N_V * N_U;
        if (p + 1 < N_PROJ)  // gfx1250 global_prefetch_b8 hint for next view
            __builtin_prefetch(pp + (size_t)N_V * N_U + cv0 * N_U + cu0, 0, 1);

        const float s00 = (vok0 && uok0) ? pp[cv0 * N_U + cu0] : 0.0f;
        const float s01 = (vok0 && uok1) ? pp[cv0 * N_U + cu1] : 0.0f;
        const float s10 = (vok1 && uok0) ? pp[cv1 * N_U + cu0] : 0.0f;
        const float s11 = (vok1 && uok1) ? pp[cv1 * N_U + cu1] : 0.0f;

        const float val = (1.0f - wv) * ((1.0f - wu) * s00 + wu * s01)
                        +         wv  * ((1.0f - wu) * s10 + wu * s11);
        acc += val * mag * mag;
    }

    acc *= 2.0f * PI_F / (float)N_PROJ;
    out[((size_t)z * VOLD + y) * VOLD + x] = fmaxf(acc, 0.0f);
}

// ---------------------------------------------------------------------------
extern "C" void kernel_launch(void* const* d_in, const int* in_sizes, int n_in,
                              void* d_out, int out_size, void* d_ws, size_t ws_size,
                              hipStream_t stream) {
    const float* sino = (const float*)d_in[0];   // (1,96,96,384) f32
    const float* wgt  = (const float*)d_in[1];   // (96,96,384)   f32
    const float* w3d  = (const float*)d_in[2];   // (384,384)     f32

    float* g_ws    = (float*)d_ws;                                        // 14.2 MB
    float* proj_ws = (float*)((char*)d_ws + (size_t)N_PROJ * N_ANG * N_DET * sizeof(float));

    // Stage 1: banded WMMA gradient. 288 row-pairs x 24 col-tiles = 6912 waves.
    grad_wmma_kernel<<<6912 / 8, 256, 0, stream>>>(sino, wgt, g_ws);

    // Stage 2: 2-D parallel backprojection (+ weight_3d), async-LDS staged.
    dim3 gridC(N_U / 16, N_V / 16, N_PROJ);      // 24 x 24 x 96
    bp2d_kernel<<<gridC, 256, 0, stream>>>(g_ws, w3d, proj_ws);

    // Stage 3: 3-D cone backprojection (+ ReLU).
    dim3 gridD(VOLD / 16, VOLD / 16, VOLD);      // 12 x 12 x 192
    bp3d_kernel<<<gridD, 256, 0, stream>>>(proj_ws, (float*)d_out);
}
